// VectorQuantizer_51024211476941
// MI455X (gfx1250) — matrix operand order repaired and checked by reading, not compiled, not run on hardware
//
#include <hip/hip_runtime.h>
#include <hip/hip_bf16.h>

// ---------------------------------------------------------------------------
// VQ-VAE vector quantizer for MI455X (gfx1250, wave32, WMMA).
//   N = 65536 inputs, D = 256, K = 4096 codes.
//   argmin_n ( |e_n|^2 - 2 x.e_n )  via v_wmma_f32_16x16x32_bf16
//   loss = 1.25 * mean((x - q)^2)   recomputed exactly in fp32 on gather
//   Double-buffered LDS staging via global_load_async_to_lds_b128 (ASYNCcnt),
//   DMA of chunk c+1 overlapped with WMMA compute of chunk c.
// ---------------------------------------------------------------------------

#define NUM_EMB   4096
#define EMB_DIM   256
#define N_IN      65536
#define CHUNK     64                 // codes staged per LDS chunk
#define LDS_STR   72                 // bf16 row stride (144B, 16B aligned, low bank conflict)
#define NCHUNKS   (NUM_EMB / CHUNK)  // 64

typedef __attribute__((ext_vector_type(16))) __bf16 v16bf;
typedef __attribute__((ext_vector_type(8)))  __bf16 v8bf;
typedef __attribute__((ext_vector_type(8)))  float  v8f;

__device__ __forceinline__ unsigned short f2bf_bits(float f) {
    __bf16 b = (__bf16)f;
    return __builtin_bit_cast(unsigned short, b);
}

__device__ __forceinline__ void wait_asynccnt0() {
#if __has_builtin(__builtin_amdgcn_s_wait_asynccnt)
    __builtin_amdgcn_s_wait_asynccnt(0);
#else
    asm volatile("s_wait_asynccnt 0x0" ::: "memory");
#endif
}

// ---------------------------------------------------------------------------
// Kernel 1: prep.  grid = 4096 (one block per code row), block = 256 (= D).
//  - |e_k|^2 into enorm[k]
//  - bf16 codebook in tiled-transposed layout: etile[(k/64)*16384 + d*64 + (k%64)]
//  - zero counts histogram
// ---------------------------------------------------------------------------
__global__ __launch_bounds__(256)
void vq_prep_kernel(const float* __restrict__ cb,
                    unsigned short* __restrict__ etile,
                    float* __restrict__ enorm,
                    int* __restrict__ counts)
{
    const int k = blockIdx.x;       // code row
    const int t = threadIdx.x;      // dim
    const float v = cb[k * EMB_DIM + t];

    etile[(k >> 6) * (EMB_DIM * CHUNK) + t * CHUNK + (k & 63)] = f2bf_bits(v);

    __shared__ float red[256];
    red[t] = v * v;
    __syncthreads();
    for (int s = 128; s > 0; s >>= 1) {
        if (t < s) red[t] += red[t + s];
        __syncthreads();
    }
    if (t == 0) enorm[k] = red[0];

    if (k < NUM_EMB / 256) counts[k * 256 + t] = 0;
}

// ---------------------------------------------------------------------------
// Kernel 2: main GEMM + argmin + gather.  grid = 512, block = 256 (8 waves).
// Each wave: 16 rows of X, all 4096 codes, 2048 bf16 WMMAs.
// ---------------------------------------------------------------------------
__global__ __launch_bounds__(256)
void vq_main_kernel(const float* __restrict__ x,
                    const float* __restrict__ cb,
                    const unsigned short* __restrict__ etile,
                    const float* __restrict__ enorm,
                    int* __restrict__ counts,
                    float* __restrict__ lossPart,
                    float* __restrict__ qout,
                    float* __restrict__ idxout)
{
    __shared__ unsigned short lds_et[2][EMB_DIM * LDS_STR];  // 2 x 36,864 B
    __shared__ float sMin[8][16];
    __shared__ int   sIdx[8][16];
    __shared__ float sLoss[8];

    const int tid  = threadIdx.x;
    const int wave = tid >> 5;
    const int lane = tid & 31;
    const int col  = lane & 15;
    const int hf   = lane >> 4;

    const int rowBase = blockIdx.x * 128 + wave * 16;
    const int myRow   = rowBase + col;

    // ---- A fragments: 16 rows x 256 dims, bf16, ISA 16-bit A layout ----
    // lane (hf=0): K = kk*32 + {0..7, 16..23};  lane (hf=1): K = kk*32 + {8..15, 24..31}
    const float* xr = x + (size_t)myRow * EMB_DIM;
    v16bf a[8];
#pragma unroll
    for (int kk = 0; kk < 8; ++kk) {
        const int b1 = kk * 32 + hf * 8;
        const int b2 = b1 + 16;
        const float4 f0 = *reinterpret_cast<const float4*>(xr + b1);
        const float4 f1 = *reinterpret_cast<const float4*>(xr + b1 + 4);
        const float4 f2 = *reinterpret_cast<const float4*>(xr + b2);
        const float4 f3 = *reinterpret_cast<const float4*>(xr + b2 + 4);
        const float tmp[16] = { f0.x, f0.y, f0.z, f0.w,  f1.x, f1.y, f1.z, f1.w,
                                f2.x, f2.y, f2.z, f2.w,  f3.x, f3.y, f3.z, f3.w };
#pragma unroll
        for (int i = 0; i < 16; ++i) a[kk][i] = (__bf16)tmp[i];
    }

    float vmin[8];
    int   vidx[8];
#pragma unroll
    for (int j = 0; j < 8; ++j) { vmin[j] = 3.0e38f; vidx[j] = 0; }

    // Per-thread async staging addresses: thread t owns dim-row d = t
    // (128B contiguous in the tiled global layout -> 144B-strided LDS row).
    const unsigned ldsRowB[2] = {
        (unsigned)(uintptr_t)&lds_et[0][tid * LDS_STR],
        (unsigned)(uintptr_t)&lds_et[1][tid * LDS_STR]
    };
    const unsigned gOfsB = (unsigned)(tid * (CHUNK * 2));   // bytes within chunk

    // Issue 8 async b128 loads for chunk c into LDS buffer `buf`.
    // Instruction offset advances both global and LDS addresses (ISA 10.x).
    auto issue_chunk = [&](int c, int buf) {
        const unsigned short* cbase = etile + (size_t)c * (EMB_DIM * CHUNK);
        const unsigned ldsb = ldsRowB[buf];
#define VQ_ASYNC_LD(IMM)                                                      \
        asm volatile("global_load_async_to_lds_b128 %0, %1, %2 offset:" #IMM  \
                     :: "v"(ldsb), "v"(gOfsB), "s"(cbase) : "memory")
        VQ_ASYNC_LD(0);  VQ_ASYNC_LD(16); VQ_ASYNC_LD(32); VQ_ASYNC_LD(48);
        VQ_ASYNC_LD(64); VQ_ASYNC_LD(80); VQ_ASYNC_LD(96); VQ_ASYNC_LD(112);
#undef VQ_ASYNC_LD
    };

    // ---- software pipeline: DMA chunk c+1 under compute of chunk c ----
    issue_chunk(0, 0);
    wait_asynccnt0();
    __syncthreads();

    for (int c = 0; c < NCHUNKS; ++c) {
        const int buf = c & 1;
        if (c + 1 < NCHUNKS) issue_chunk(c + 1, buf ^ 1);
        if (c + 2 < NCHUNKS)
            __builtin_prefetch(etile + (size_t)(c + 2) * (EMB_DIM * CHUNK) + tid * CHUNK, 0, 0);

        const unsigned short* lbuf = &lds_et[buf][0];

#pragma unroll
        for (int tn = 0; tn < 4; ++tn) {
            const int n0 = c * CHUNK + tn * 16;
            v8f acc0 = {};
            v8f acc1 = {};
#pragma unroll
            for (int kk = 0; kk < 8; kk += 2) {
                {   // B fragment: lane = N column (tn*16 + col), K = kk*32 + hf*8 + {0..7, 16..23}
                    const unsigned short* p =
                        lbuf + (kk * 32 + hf * 8) * LDS_STR + tn * 16 + col;
                    v16bf b;
#pragma unroll
                    for (int i = 0; i < 8; ++i) { b[i] = __builtin_bit_cast(__bf16, p[i * LDS_STR]); b[8 + i] = __builtin_bit_cast(__bf16, p[(16 + i) * LDS_STR]); }
                    acc0 = __builtin_amdgcn_wmma_f32_16x16x32_bf16(
                        false, a[kk], false, b, (short)0, acc0, false, false);
                }
                {
                    const unsigned short* p =
                        lbuf + ((kk + 1) * 32 + hf * 8) * LDS_STR + tn * 16 + col;
                    v16bf b;
#pragma unroll
                    for (int i = 0; i < 8; ++i) { b[i] = __builtin_bit_cast(__bf16, p[i * LDS_STR]); b[8 + i] = __builtin_bit_cast(__bf16, p[(16 + i) * LDS_STR]); }
                    acc1 = __builtin_amdgcn_wmma_f32_16x16x32_bf16(
                        false, a[kk + 1], false, b, (short)0, acc1, false, false);
                }
            }
            // C/D layout: slot (VGPR j, lane) = row j + 8*hf, code n0 + col
            const float en  = enorm[n0 + col];
            const int   nId = n0 + col;
#pragma unroll
            for (int j = 0; j < 8; ++j) {
                const float val = fmaf(-2.0f, acc0[j] + acc1[j], en);
                if (val < vmin[j]) { vmin[j] = val; vidx[j] = nId; }
            }
        }

        wait_asynccnt0();   // chunk c+1 landed (own loads)
        __syncthreads();    // all waves done with chunk c, chunk c+1 visible
    }

    // ---- argmin reduce across the 16 lane-columns (xor stays in 16-group) ----
#pragma unroll
    for (int j = 0; j < 8; ++j) {
        float m  = vmin[j];
        int   id = vidx[j];
#pragma unroll
        for (int off = 1; off < 16; off <<= 1) {
            const float om = __shfl_xor(m, off, 32);
            const int   oi = __shfl_xor(id, off, 32);
            if (om < m || (om == m && oi < id)) { m = om; id = oi; }
        }
        vmin[j] = m; vidx[j] = id;
    }
    if (col == 0) {
#pragma unroll
        for (int j = 0; j < 8; ++j) {
            sMin[wave][hf * 8 + j] = vmin[j];
            sIdx[wave][hf * 8 + j] = vidx[j];
        }
    }
    __syncthreads();

    // ---- indices + histogram ----
    if (lane < 16) {
        const int idx = sIdx[wave][lane];
        idxout[rowBase + lane] = (float)idx;
        atomicAdd(&counts[idx], 1);
    }

    // ---- gather quantized rows + exact fp32 loss ----
    float lacc = 0.f;
    for (int r = 0; r < 16; ++r) {
        const int idx = sIdx[wave][r];
        const float4* crow = reinterpret_cast<const float4*>(cb + (size_t)idx * EMB_DIM);
        const float4* xrow = reinterpret_cast<const float4*>(x  + (size_t)(rowBase + r) * EMB_DIM);
        float4*       orow = reinterpret_cast<float4*>(qout + (size_t)(rowBase + r) * EMB_DIM);
#pragma unroll
        for (int i = 0; i < 2; ++i) {
            const int e = lane + i * 32;
            const float4 cv = crow[e];
            const float4 xv = xrow[e];
            const float dx = xv.x - cv.x, dy = xv.y - cv.y;
            const float dz = xv.z - cv.z, dw = xv.w - cv.w;
            lacc += dx * dx + dy * dy + dz * dz + dw * dw;
            orow[e] = cv;
        }
    }
#pragma unroll
    for (int off = 16; off > 0; off >>= 1) lacc += __shfl_xor(lacc, off, 32);
    if (lane == 0) sLoss[wave] = lacc;
    __syncthreads();
    if (tid == 0) {
        float s = 0.f;
        for (int wv = 0; wv < 8; ++wv) s += sLoss[wv];
        lossPart[blockIdx.x] = s;          // plain store: deterministic
    }
}

// ---------------------------------------------------------------------------
// Kernel 3: finalize.  1 block x 256 threads, deterministic reduction order.
// ---------------------------------------------------------------------------
__global__ __launch_bounds__(256)
void vq_finalize_kernel(const float* __restrict__ lossPart,
                        const int* __restrict__ counts,
                        float* __restrict__ outScalars)
{
    __shared__ float rl[256];
    __shared__ float rp[256];
    const int t = threadIdx.x;

    float s = 0.f;
    for (int i = t; i < 512; i += 256) s += lossPart[i];
    float e = 0.f;
    for (int i = t; i < NUM_EMB; i += 256) {
        const float p = (float)counts[i] * (1.0f / (float)N_IN);
        e += p * logf(p + 1e-10f);
    }
    rl[t] = s; rp[t] = e;
    __syncthreads();
    for (int st = 128; st > 0; st >>= 1) {
        if (t < st) { rl[t] += rl[t + st]; rp[t] += rp[t + st]; }
        __syncthreads();
    }
    if (t == 0) {
        outScalars[0] = 1.25f * rl[0] / (float)((size_t)N_IN * EMB_DIM);  // loss
        outScalars[1] = expf(-rp[0]);                                     // perplexity
    }
}

// ---------------------------------------------------------------------------
// Launch.  Workspace layout (bytes):
//   [0, 2097152)        etile bf16 (tiled transposed codebook)
//   [2097152, 2113536)  enorm f32[4096]
//   [2113536, 2129920)  counts i32[4096]
//   [2129920, 2131968)  lossPart f32[512]
// d_out (float):  quantized[N*D] | loss | perplexity | indices[N]
// ---------------------------------------------------------------------------
extern "C" void kernel_launch(void* const* d_in, const int* in_sizes, int n_in,
                              void* d_out, int out_size, void* d_ws, size_t ws_size,
                              hipStream_t stream)
{
    (void)in_sizes; (void)n_in; (void)out_size; (void)ws_size;

    const float* x  = (const float*)d_in[0];   // [65536, 256] f32
    const float* cb = (const float*)d_in[1];   // [4096, 256]  f32

    char* ws = (char*)d_ws;
    unsigned short* etile    = (unsigned short*)(ws);
    float*          enorm    = (float*)(ws + 2097152);
    int*            counts   = (int*)  (ws + 2113536);
    float*          lossPart = (float*)(ws + 2129920);

    float* out    = (float*)d_out;
    float* qout   = out;
    float* scal   = out + (size_t)N_IN * EMB_DIM;  // [loss, perplexity]
    float* idxout = scal + 2;                      // indices as float

    vq_prep_kernel<<<NUM_EMB, 256, 0, stream>>>(cb, etile, enorm, counts);
    vq_main_kernel<<<N_IN / 128, 256, 0, stream>>>(x, cb, etile, enorm, counts,
                                                   lossPart, qout, idxout);
    vq_finalize_kernel<<<1, 256, 0, stream>>>(lossPart, counts, scal);
}
